// MolecularRGCN_33586644254847
// MI455X (gfx1250) — matrix operand order repaired and checked by reading, not compiled, not run on hardware
//
#include <hip/hip_runtime.h>

#define F       128
#define R       4
#define NNODES  50000
#define NEDGES  1600000
#define OUTF    1024
#define LAYERS  3
#define MT      5            // row tiles (of 16) per block -> 80 rows/block
#define ROWS    (MT * 16)
#define NTILEBLK (NNODES / ROWS)   // 625

// Padded LDS strides (halves): 136 halves = 272 B = 68 dwords -> bank index
// advances by 4 per row/col, so b128 fragment reads are conflict-free.
#define LDA     136
#define LDB     136

typedef __attribute__((ext_vector_type(16))) _Float16 v16h;
typedef __attribute__((ext_vector_type(8)))  _Float16 v8h;
typedef __attribute__((ext_vector_type(8)))  float    v8f;

// ---------------------------------------------------------------------------
// Utility kernels
// ---------------------------------------------------------------------------
__global__ void k_zero(float* __restrict__ p, int n) {
    int i = blockIdx.x * blockDim.x + threadIdx.x;
    if (i < n) p[i] = 0.0f;
}

__global__ void k_count(const int* __restrict__ dst, const int* __restrict__ et,
                        float* __restrict__ cnt) {
    int e = blockIdx.x * blockDim.x + threadIdx.x;
    if (e < NEDGES) atomicAdd(&cnt[dst[e] * R + et[e]], 1.0f);
}

// ---------------------------------------------------------------------------
// WMMA fragment helpers (CDNA5 16-bit A 16x32 / B 32x16 layouts, wave32).
// All LDS reads are aligned 16-byte v8h loads -> ds_load_b128.
// ---------------------------------------------------------------------------
__device__ __forceinline__ v16h cat16(v8h lo, v8h hi) {
    return __builtin_shufflevector(lo, hi, 0, 1, 2, 3, 4, 5, 6, 7,
                                   8, 9, 10, 11, 12, 13, 14, 15);
}

// A halves: k = kbase + {0..7} + hi*8  and  k = kbase + 16 + {0..7} + hi*8
__device__ __forceinline__ v16h frag_a(const _Float16* __restrict__ ap, int kbase) {
    const v8h a0 = *(const v8h*)(ap + kbase);
    const v8h a1 = *(const v8h*)(ap + kbase + 16);
    return cat16(a0, a1);
}

// B halves: k = kbase + {0..7} + hi*8  and  k = kbase + 16 + {0..7} + hi*8
__device__ __forceinline__ v16h frag_b(const _Float16* __restrict__ bp, int kbase) {
    const v8h b0 = *(const v8h*)(bp + kbase);
    const v8h b1 = *(const v8h*)(bp + kbase + 16);
    return cat16(b0, b1);
}

// Wave computes MT stacked 16x16 C tiles (80 x 16 strip), K = 128.
// B fragment is loaded once per K-step and reused across the 5 row tiles.
__device__ __forceinline__ void wave_gemm_mt(const _Float16* __restrict__ As,
                                             const _Float16* __restrict__ Bsc,
                                             int mn, int hi, int col,
                                             v8f acc[MT]) {
    const _Float16* ap = As + mn * LDA + hi * 8;
    const _Float16* bp = Bsc + col * LDB + hi * 8;
#pragma unroll
    for (int kk = 0; kk < 4; ++kk) {
        const int kbase = kk * 32;
        const v16h b = frag_b(bp, kbase);
#pragma unroll
        for (int mt = 0; mt < MT; ++mt) {
            const v16h a = frag_a(ap + mt * 16 * LDA, kbase);
            acc[mt] = __builtin_amdgcn_wmma_f32_16x16x32_f16(
                    false, a, false, b, (short)0, acc[mt], false, false);
        }
    }
}

// ---------------------------------------------------------------------------
// Per-layer dense transform: blockIdx.y = r in [0,R): h[r] = x @ W_r
//                            blockIdx.y = R:          agg  = x @ root + bias
// Block = 256 threads (8 wave32s); block covers 80 rows; wave w -> cols
// [16w, 16w+16). ReLU of the previous layer is fused into A staging.
// ---------------------------------------------------------------------------
__global__ __launch_bounds__(256) void k_transform(
        const float* __restrict__ x, const float* __restrict__ Wrel,
        const float* __restrict__ root, const float* __restrict__ bias,
        float* __restrict__ h, float* __restrict__ agg, int applyRelu) {
    __shared__ _Float16 Bsc[F * LDB];     // column-major weight panel (f16)
    __shared__ _Float16 As[ROWS * LDA];   // 80-row node tile (f16)

    const int blk = blockIdx.x;
    const int y   = blockIdx.y;
    const float* W   = (y < R) ? (Wrel + (size_t)y * F * F) : root;
    float*       out = (y < R) ? (h + (size_t)y * NNODES * F) : agg;

    for (int i = threadIdx.x; i < F * F; i += 256) {
        const int k = i >> 7, c = i & 127;            // W row-major [k][c]
        Bsc[c * LDB + k] = (_Float16)W[i];            // store column-major
    }
    for (int i = threadIdx.x; i < ROWS * F; i += 256) {
        const int r = i >> 7, c = i & 127;
        float v = x[(size_t)blk * ROWS * F + i];
        if (applyRelu) v = fmaxf(v, 0.0f);            // fused ReLU of prev layer
        As[r * LDA + c] = (_Float16)v;
    }
    __syncthreads();

    const int lane    = threadIdx.x & 31;
    const int wave    = threadIdx.x >> 5;
    const int colbase = wave * 16;
    const int hi      = lane >> 4;
    const int mn      = lane & 15;
    const int col     = colbase + mn;

    v8f acc[MT] = {};
    wave_gemm_mt(As, Bsc, mn, hi, col, acc);

    const float bv = (y == R) ? bias[col] : 0.0f;
#pragma unroll
    for (int mt = 0; mt < MT; ++mt) {
#pragma unroll
        for (int v = 0; v < 8; ++v) {
            const int row = blk * ROWS + mt * 16 + v + hi * 8;  // M = v + 8*hi
            out[(size_t)row * F + col] = acc[mt][v] + bv;
        }
    }
}

// ---------------------------------------------------------------------------
// Edge scatter: agg[dst] += h[etype][src] / cnt[dst*R+etype]
// 32 lanes per edge, float4 message loads, f32 global atomics (working set
// h + x buffers ~155 MB fits the 192 MB L2 -> atomics run at L2 rate).
// ---------------------------------------------------------------------------
__global__ __launch_bounds__(256) void k_scatter(
        const float* __restrict__ h, const int* __restrict__ src,
        const int* __restrict__ dst, const int* __restrict__ et,
        const float* __restrict__ cnt, float* __restrict__ agg) {
    const long long t = (long long)blockIdx.x * blockDim.x + threadIdx.x;
    const int e = (int)(t >> 5);
    if (e >= NEDGES) return;
    const int lane = (int)(t & 31);

    const int s = src[e], d = dst[e], r = et[e];
    const float inv = 1.0f / cnt[d * R + r];

    const float4 m = *(const float4*)(h + ((size_t)r * NNODES + s) * F + lane * 4);
    float* ap = agg + (size_t)d * F + lane * 4;
    atomicAdd(ap + 0, m.x * inv);
    atomicAdd(ap + 1, m.y * inv);
    atomicAdd(ap + 2, m.z * inv);
    atomicAdd(ap + 3, m.w * inv);
}

// ---------------------------------------------------------------------------
// Final FC: out[N,1024] = relu(x) @ fc_w + fc_b   (ReLU of layer 3 fused)
// ---------------------------------------------------------------------------
__global__ __launch_bounds__(256) void k_fc(
        const float* __restrict__ x, const float* __restrict__ w,
        const float* __restrict__ b, float* __restrict__ out) {
    __shared__ _Float16 Bsc[F * LDB];
    __shared__ _Float16 As[ROWS * LDA];

    const int blk = blockIdx.x;
    const int cg  = blockIdx.y;                       // 128-col output group

    for (int i = threadIdx.x; i < F * F; i += 256) {
        const int k = i >> 7, c = i & 127;
        Bsc[c * LDB + k] = (_Float16)w[(size_t)k * OUTF + cg * F + c];
    }
    for (int i = threadIdx.x; i < ROWS * F; i += 256) {
        const int r = i >> 7, c = i & 127;
        As[r * LDA + c] = (_Float16)fmaxf(x[(size_t)blk * ROWS * F + i], 0.0f);
    }
    __syncthreads();

    const int lane    = threadIdx.x & 31;
    const int wave    = threadIdx.x >> 5;
    const int colbase = wave * 16;
    const int hi      = lane >> 4;
    const int mn      = lane & 15;

    v8f acc[MT] = {};
    wave_gemm_mt(As, Bsc, mn, hi, colbase + mn, acc);

    const int col = cg * F + colbase + mn;
    const float bv = b[col];
#pragma unroll
    for (int mt = 0; mt < MT; ++mt) {
#pragma unroll
        for (int v = 0; v < 8; ++v) {
            const int row = blk * ROWS + mt * 16 + v + hi * 8;
            out[(size_t)row * OUTF + col] = acc[mt][v] + bv;
        }
    }
}

// ---------------------------------------------------------------------------
// Host-side launch
// ---------------------------------------------------------------------------
extern "C" void kernel_launch(void* const* d_in, const int* in_sizes, int n_in,
                              void* d_out, int out_size, void* d_ws, size_t ws_size,
                              hipStream_t stream) {
    const float* x0     = (const float*)d_in[0];
    const int*   eidx   = (const int*)d_in[1];
    const int*   etype  = (const int*)d_in[2];
    const float* Ws     = (const float*)d_in[3];   // [L,R,F,F]
    const float* roots  = (const float*)d_in[4];   // [L,F,F]
    const float* biases = (const float*)d_in[5];   // [L,F]
    const float* fc_w   = (const float*)d_in[6];   // [F,OUT]
    const float* fc_b   = (const float*)d_in[7];   // [OUT]
    float* out = (float*)d_out;

    const int* src = eidx;           // edge_index[0]
    const int* dst = eidx + NEDGES;  // edge_index[1]

    // Workspace layout (floats): cnt | h | xbuf0 | xbuf1   (~155 MB total)
    float* ws  = (float*)d_ws;
    float* cnt = ws;                                   // N*R
    float* h   = cnt + (size_t)NNODES * R;             // R*N*F
    float* xb0 = h + (size_t)R * NNODES * F;           // N*F
    float* xb1 = xb0 + (size_t)NNODES * F;             // N*F

    k_zero<<<(NNODES * R + 255) / 256, 256, 0, stream>>>(cnt, NNODES * R);
    k_count<<<(NEDGES + 255) / 256, 256, 0, stream>>>(dst, etype, cnt);

    const float* xin = x0;
    float* bufs[2] = { xb0, xb1 };
    for (int l = 0; l < LAYERS; ++l) {
        float* agg = bufs[l & 1];
        k_transform<<<dim3(NTILEBLK, R + 1), 256, 0, stream>>>(
            xin, Ws + (size_t)l * R * F * F, roots + (size_t)l * F * F,
            biases + (size_t)l * F, h, agg, l > 0 ? 1 : 0);

        const long long sthreads = (long long)NEDGES * 32;
        k_scatter<<<(int)((sthreads + 255) / 256), 256, 0, stream>>>(
            h, src, dst, etype, cnt, agg);
        xin = agg;   // pre-ReLU; ReLU fused into next stage's A conversion
    }

    k_fc<<<dim3(NTILEBLK, OUTF / F), 256, 0, stream>>>(xin, fc_w, fc_b, out);
}
